// LightweightGATEncoder_704374636750
// MI455X (gfx1250) — compile-verified, matrix operand order checked by reading, not run on hardware
//
#include <hip/hip_runtime.h>
#include <hip/hip_bf16.h>

typedef __attribute__((ext_vector_type(16))) _Float16 v16h;
typedef __attribute__((ext_vector_type(8)))  _Float16 v8h;
typedef __attribute__((ext_vector_type(8)))  float    v8f;

#define HID 32
#define HEADS 2
#define FUSE_IN 99
#define XC_STRIDE 100
#define KP1 608          // ceil(99*6 / 32)*32
#define KP2 192          // 32*6
#define XES 608          // LDS row stride (halfs) for expanded KAN input
#define NEG_SLOPE 0.2f

// ---------- helpers ----------
__device__ __forceinline__ float lrelu(float v) { return v > 0.f ? v : NEG_SLOPE * v; }

// monotonic float<->uint mapping for atomic max on floats
__device__ __forceinline__ unsigned f2ord(float f) {
    unsigned u = __float_as_uint(f);
    return (u & 0x80000000u) ? ~u : (u | 0x80000000u);
}
__device__ __forceinline__ float ord2f(unsigned u) {
    return __uint_as_float((u & 0x80000000u) ? (u & 0x7fffffffu) : ~u);
}

// order-2 (K=2) B-spline bases, 8 grid points -> 5 basis values (Cox-de Boor)
__device__ __forceinline__ void bspline5(float x, const float* g8, float* out5) {
    float b0[7];
#pragma unroll
    for (int t = 0; t < 7; ++t)
        b0[t] = (x >= g8[t] && x < g8[t + 1]) ? 1.f : 0.f;
    float b1[6];
#pragma unroll
    for (int t = 0; t < 6; ++t)
        b1[t] = (x - g8[t]) / (g8[t + 1] - g8[t]) * b0[t]
              + (g8[t + 2] - x) / (g8[t + 2] - g8[t + 1]) * b0[t + 1];
#pragma unroll
    for (int t = 0; t < 5; ++t)
        out5[t] = (x - g8[t]) / (g8[t + 2] - g8[t]) * b1[t]
                + (g8[t + 3] - x) / (g8[t + 3] - g8[t + 1]) * b1[t + 1];
}

__device__ __forceinline__ v16h cat8(v8h lo, v8h hi) {
    return __builtin_shufflevector(lo, hi, 0, 1, 2, 3, 4, 5, 6, 7,
                                   8, 9, 10, 11, 12, 13, 14, 15);
}

// ---------- 0) f32 -> f16 copy ----------
__global__ __launch_bounds__(256)
void f32_to_f16_kernel(const float* __restrict__ src, _Float16* __restrict__ dst, int n) {
    int i = blockIdx.x * blockDim.x + threadIdx.x;
    if (i < n) dst[i] = (_Float16)src[i];
}

// ---------- 1) xp = x @ W^T  (WMMA f16 -> f32) ----------
// block: 128 threads = 4 waves; wave w owns output cols [w*16, w*16+16)
__global__ __launch_bounds__(128)
void gemm_xp_kernel(const float* __restrict__ x, const _Float16* __restrict__ Wh,
                    float* __restrict__ xp, int N) {
    __shared__ __align__(16) _Float16 sx[16 * 64];
    __shared__ __align__(16) _Float16 sw[64 * 64];
    const int tid = threadIdx.x;
    const int node0 = blockIdx.x * 16;

    {   // x tile: 1024 floats, 8 consecutive per thread, converted to f16
        int i = tid * 8;                       // exactly one chunk per thread
        int r = i >> 6, k = i & 63;
        int n = node0 + r;
        if (n < N) {
            const float* px = &x[(size_t)n * 64 + k];
#pragma unroll
            for (int j = 0; j < 8; ++j) sx[i + j] = (_Float16)px[j];
        } else {
#pragma unroll
            for (int j = 0; j < 8; ++j) sx[i + j] = (_Float16)0.f;
        }
    }
    for (int i = tid * 8; i < 64 * 64; i += 128 * 8)   // W tile: b128 copies
        *(v8h*)&sw[i] = *(const v8h*)&Wh[i];
    __syncthreads();

    const int wv = tid >> 5;          // output col tile
    const int lane = tid & 31;
    const int mr = lane & 15;
    const int hi = lane >> 4;
    v8f c = {};
#pragma unroll
    for (int s = 0; s < 2; ++s) {
        const _Float16* pa = &sx[mr * 64 + s * 32 + hi * 8];
        const _Float16* pb = &sw[(wv * 16 + mr) * 64 + s * 32 + hi * 8];
        v16h a = cat8(*(const v8h*)pa, *(const v8h*)(pa + 16));
        v16h b = cat8(*(const v8h*)pb, *(const v8h*)(pb + 16));
        c = __builtin_amdgcn_wmma_f32_16x16x32_f16(false, a, false, b, (short)0, c,
                                                   false, false);
    }
    if (node0 + 16 <= N) {            // uniform fast path
#pragma unroll
        for (int r = 0; r < 8; ++r)
            xp[(size_t)(node0 + hi * 8 + r) * 64 + wv * 16 + mr] = c[r];
    } else {
#pragma unroll
        for (int r = 0; r < 8; ++r) {
            int n = node0 + hi * 8 + r;
            if (n < N) xp[(size_t)n * 64 + wv * 16 + mr] = c[r];
        }
    }
}

// ---------- 2) per-node attention logits ----------
__global__ __launch_bounds__(256)
void logits_kernel(const float* __restrict__ xp, const float* __restrict__ att_s,
                   const float* __restrict__ att_d, float* __restrict__ a_s,
                   float* __restrict__ a_d, int N) {
    int n = blockIdx.x * blockDim.x + threadIdx.x;
    if (n >= N) return;
#pragma unroll
    for (int h = 0; h < HEADS; ++h) {
        float ss = 0.f, dd = 0.f;
#pragma unroll
        for (int cc = 0; cc < HID; ++cc) {
            float v = xp[(size_t)n * 64 + h * HID + cc];
            ss += v * att_s[h * HID + cc];
            dd += v * att_d[h * HID + cc];
        }
        a_s[n * HEADS + h] = ss;
        a_d[n * HEADS + h] = dd;
    }
}

// ---------- 3) segment-softmax passes ----------
__global__ __launch_bounds__(256)
void init_max_kernel(const float* __restrict__ a_s, const float* __restrict__ a_d,
                     unsigned* __restrict__ maxu, int N) {
    int n = blockIdx.x * blockDim.x + threadIdx.x;
    if (n >= N) return;
#pragma unroll
    for (int h = 0; h < HEADS; ++h)
        maxu[n * HEADS + h] = f2ord(lrelu(a_s[n * HEADS + h] + a_d[n * HEADS + h]));
}

__global__ __launch_bounds__(256)
void edge_max_kernel(const int* __restrict__ ei, const float* __restrict__ a_s,
                     const float* __restrict__ a_d, unsigned* __restrict__ maxu, int E) {
    int e = blockIdx.x * blockDim.x + threadIdx.x;
    if (e >= E) return;
    int s = ei[e], d = ei[E + e];
#pragma unroll
    for (int h = 0; h < HEADS; ++h) {
        float al = lrelu(a_s[s * HEADS + h] + a_d[d * HEADS + h]);
        atomicMax(&maxu[d * HEADS + h], f2ord(al));
    }
}

__global__ __launch_bounds__(256)
void node_denom_kernel(const float* __restrict__ a_s, const float* __restrict__ a_d,
                       const unsigned* __restrict__ maxu, float* __restrict__ denom, int N) {
    int n = blockIdx.x * blockDim.x + threadIdx.x;
    if (n >= N) return;
#pragma unroll
    for (int h = 0; h < HEADS; ++h) {
        float m = ord2f(maxu[n * HEADS + h]);
        denom[n * HEADS + h] =
            __expf(lrelu(a_s[n * HEADS + h] + a_d[n * HEADS + h]) - m);
    }
}

__global__ __launch_bounds__(256)
void edge_denom_kernel(const int* __restrict__ ei, const float* __restrict__ a_s,
                       const float* __restrict__ a_d, const unsigned* __restrict__ maxu,
                       float* __restrict__ denom, int E) {
    int e = blockIdx.x * blockDim.x + threadIdx.x;
    if (e >= E) return;
    int s = ei[e], d = ei[E + e];
#pragma unroll
    for (int h = 0; h < HEADS; ++h) {
        float al = lrelu(a_s[s * HEADS + h] + a_d[d * HEADS + h]);
        float m = ord2f(maxu[d * HEADS + h]);
        atomicAdd(&denom[d * HEADS + h], __expf(al - m));
    }
}

// self-loop message: thread per (node, channel) -> fully coalesced init of agg
__global__ __launch_bounds__(256)
void node_selfmsg_kernel(const float* __restrict__ xp, const float* __restrict__ a_s,
                         const float* __restrict__ a_d, const unsigned* __restrict__ maxu,
                         const float* __restrict__ denom, float* __restrict__ agg, int N) {
    int idx = blockIdx.x * blockDim.x + threadIdx.x;
    if (idx >= N * 64) return;
    int n = idx >> 6, cc = idx & 63, h = cc >> 5;
    float m = ord2f(maxu[n * HEADS + h]);
    float coef = __expf(lrelu(a_s[n * HEADS + h] + a_d[n * HEADS + h]) - m) /
                 denom[n * HEADS + h];
    agg[(size_t)n * 64 + cc] = xp[(size_t)n * 64 + cc] * coef;
}

// edge messages: 32 lanes per edge, each lane owns channels lane and lane+32 ->
// every wave issues atomics to 32 consecutive floats of one dst row (coalesced),
// and the xp[src] gather is coalesced the same way.
__global__ __launch_bounds__(256)
void edge_msg_kernel(const int* __restrict__ ei, const float* __restrict__ xp,
                     const float* __restrict__ a_s, const float* __restrict__ a_d,
                     const unsigned* __restrict__ maxu, const float* __restrict__ denom,
                     float* __restrict__ agg, int E) {
    int idx = blockIdx.x * blockDim.x + threadIdx.x;
    int e = idx >> 5;
    int lane = idx & 31;
    if (e >= E) return;
    int s = ei[e], d = ei[E + e];
#pragma unroll
    for (int h = 0; h < HEADS; ++h) {
        float al = lrelu(a_s[s * HEADS + h] + a_d[d * HEADS + h]);
        float m = ord2f(maxu[d * HEADS + h]);
        float coef = __expf(al - m) / denom[d * HEADS + h];
        int cc = h * HID + lane;
        atomicAdd(&agg[(size_t)d * 64 + cc], xp[(size_t)s * 64 + cc] * coef);
    }
}

// ---------- 4) combine: head-mean + bias, append softmax(alpha) scalars ----------
__global__ __launch_bounds__(256)
void combine_kernel(const float* __restrict__ aggA, const float* __restrict__ aggO,
                    const float* __restrict__ aggT, const float* __restrict__ bA,
                    const float* __restrict__ bO, const float* __restrict__ bT,
                    const float* __restrict__ pa, const float* __restrict__ po,
                    const float* __restrict__ pt, float* __restrict__ xc, int N) {
    int idx = blockIdx.x * blockDim.x + threadIdx.x;
    if (idx >= N * HID) return;
    int n = idx >> 5, cc = idx & 31;
    xc[n * XC_STRIDE + cc]      = 0.5f * (aggA[(size_t)n * 64 + cc] + aggA[(size_t)n * 64 + 32 + cc]) + bA[cc];
    xc[n * XC_STRIDE + 32 + cc] = 0.5f * (aggO[(size_t)n * 64 + cc] + aggO[(size_t)n * 64 + 32 + cc]) + bO[cc];
    xc[n * XC_STRIDE + 64 + cc] = 0.5f * (aggT[(size_t)n * 64 + cc] + aggT[(size_t)n * 64 + 32 + cc]) + bT[cc];
    if (cc < 4) {
        float va = *pa, vo = *po, vt = *pt;
        float m = fmaxf(va, fmaxf(vo, vt));
        float ea = __expf(va - m), eo = __expf(vo - m), et = __expf(vt - m);
        float sum = ea + eo + et;
        float wv[4] = {ea / sum, eo / sum, et / sum, 0.f};
        xc[n * XC_STRIDE + 96 + cc] = wv[cc];
    }
}

// ---------- 5) expand KAN weights to f16 (base | 5 spline) ----------
__global__ __launch_bounds__(256)
void prep_we_kernel(const float* __restrict__ base_w, const float* __restrict__ spline_w,
                    _Float16* __restrict__ We, int in_feats, int Kp) {
    int idx = blockIdx.x * blockDim.x + threadIdx.x;
    if (idx >= HID * Kp) return;
    int o = idx / Kp, k = idx % Kp;
    float v = 0.f;
    if (k < in_feats * 6) {
        int i = k / 6, cc = k % 6;
        v = (cc == 0) ? base_w[o * in_feats + i]
                      : spline_w[(o * in_feats + i) * 5 + (cc - 1)];
    }
    We[idx] = (_Float16)v;
}

// ---------- 6) KAN layer as WMMA GEMM over expanded input ----------
// block: 64 threads = 2 waves; wave w owns output cols [w*16, w*16+16)
__global__ __launch_bounds__(64)
void kan_kernel(const float* __restrict__ xin, int xstride, int in_feats,
                const float* __restrict__ grid, const _Float16* __restrict__ We, int Kp,
                float* __restrict__ out, int relu, int N) {
    __shared__ __align__(16) _Float16 xe[16 * XES];
    const int tid = threadIdx.x;
    const int node0 = blockIdx.x * 16;
    for (int i = tid * 8; i < 16 * XES; i += 64 * 8)
        *(v8h*)&xe[i] = (v8h)(_Float16)0.f;
    __syncthreads();
    for (int it = tid; it < 16 * in_feats; it += 64) {
        int r = it / in_feats, i = it % in_feats;
        int n = node0 + r;
        float xv = (n < N) ? xin[(size_t)n * xstride + i] : 0.f;
        float sl = xv / (1.f + __expf(-xv));      // silu
        float g8[8];
#pragma unroll
        for (int j = 0; j < 8; ++j) g8[j] = grid[i * 8 + j];
        float b5[5];
        bspline5(xv, g8, b5);
        _Float16* p = &xe[r * XES + i * 6];
        p[0] = (_Float16)sl;
#pragma unroll
        for (int j = 0; j < 5; ++j) p[1 + j] = (_Float16)b5[j];
    }
    __syncthreads();

    const int wv = tid >> 5;
    const int lane = tid & 31;
    const int mr = lane & 15;
    const int hi = lane >> 4;
    v8f c = {};
    for (int s = 0; s < Kp / 32; ++s) {
        const _Float16* pa = &xe[mr * XES + s * 32 + hi * 8];
        const _Float16* pb = &We[(wv * 16 + mr) * Kp + s * 32 + hi * 8];
        v16h a = cat8(*(const v8h*)pa, *(const v8h*)(pa + 16));
        v16h b = cat8(*(const v8h*)pb, *(const v8h*)(pb + 16));
        c = __builtin_amdgcn_wmma_f32_16x16x32_f16(false, a, false, b, (short)0, c,
                                                   false, false);
    }
    if (node0 + 16 <= N) {            // uniform fast path
#pragma unroll
        for (int r = 0; r < 8; ++r) {
            float v = c[r];
            if (relu) v = fmaxf(v, 0.f);
            out[(size_t)(node0 + hi * 8 + r) * HID + wv * 16 + mr] = v;
        }
    } else {
#pragma unroll
        for (int r = 0; r < 8; ++r) {
            int n = node0 + hi * 8 + r;
            if (n < N) {
                float v = c[r];
                if (relu) v = fmaxf(v, 0.f);
                out[(size_t)n * HID + wv * 16 + mr] = v;
            }
        }
    }
}

// ---------- launch ----------
static inline size_t alignUp(size_t v) { return (v + 255) & ~(size_t)255; }

extern "C" void kernel_launch(void* const* d_in, const int* in_sizes, int n_in,
                              void* d_out, int out_size, void* d_ws, size_t ws_size,
                              hipStream_t stream) {
    const float* x = (const float*)d_in[0];
    const int* ei[3] = {(const int*)d_in[1], (const int*)d_in[2], (const int*)d_in[3]};
    const float* alph[3] = {(const float*)d_in[4], (const float*)d_in[5], (const float*)d_in[6]};
    const float* Wg[3]  = {(const float*)d_in[7],  (const float*)d_in[11], (const float*)d_in[15]};
    const float* asg[3] = {(const float*)d_in[8],  (const float*)d_in[12], (const float*)d_in[16]};
    const float* adg[3] = {(const float*)d_in[9],  (const float*)d_in[13], (const float*)d_in[17]};
    const float* bg[3]  = {(const float*)d_in[10], (const float*)d_in[14], (const float*)d_in[18]};
    const float* base_w1   = (const float*)d_in[19];
    const float* spline_w1 = (const float*)d_in[20];
    const float* grid1     = (const float*)d_in[21];
    const float* base_w2   = (const float*)d_in[22];
    const float* spline_w2 = (const float*)d_in[23];
    const float* grid2     = (const float*)d_in[24];

    const int N = in_sizes[0] / 64;
    const int E = in_sizes[1] / 2;

    // workspace carve-out
    char* w = (char*)d_ws;
    float* xp    = (float*)w;  w += alignUp((size_t)3 * N * 64 * 4);
    float* agg   = (float*)w;  w += alignUp((size_t)3 * N * 64 * 4);
    float* a_s   = (float*)w;  w += alignUp((size_t)3 * N * HEADS * 4);
    float* a_d   = (float*)w;  w += alignUp((size_t)3 * N * HEADS * 4);
    unsigned* mx = (unsigned*)w; w += alignUp((size_t)3 * N * HEADS * 4);
    float* denom = (float*)w;  w += alignUp((size_t)3 * N * HEADS * 4);
    float* xc    = (float*)w;  w += alignUp((size_t)N * XC_STRIDE * 4);
    float* h1    = (float*)w;  w += alignUp((size_t)N * HID * 4);
    _Float16* We1 = (_Float16*)w; w += alignUp((size_t)HID * KP1 * 2);
    _Float16* We2 = (_Float16*)w; w += alignUp((size_t)HID * KP2 * 2);
    _Float16* Wh  = (_Float16*)w; w += alignUp((size_t)3 * 64 * 64 * 2);

    const int mtiles = (N + 15) / 16;
    const int nblkN  = (N + 255) / 256;
    const int nblkE  = (E + 255) / 256;

    // expanded KAN weights + f16 GAT weights
    prep_we_kernel<<<(HID * KP1 + 255) / 256, 256, 0, stream>>>(base_w1, spline_w1, We1,
                                                                FUSE_IN, KP1);
    prep_we_kernel<<<(HID * KP2 + 255) / 256, 256, 0, stream>>>(base_w2, spline_w2, We2,
                                                                HID, KP2);
    for (int g = 0; g < 3; ++g)
        f32_to_f16_kernel<<<(64 * 64 + 255) / 256, 256, 0, stream>>>(Wg[g],
                                                                     Wh + (size_t)g * 64 * 64,
                                                                     64 * 64);

    for (int g = 0; g < 3; ++g) {
        float* xpg = xp + (size_t)g * N * 64;
        float* agr = agg + (size_t)g * N * 64;
        float* asg_n = a_s + (size_t)g * N * HEADS;
        float* adg_n = a_d + (size_t)g * N * HEADS;
        unsigned* mxg = mx + (size_t)g * N * HEADS;
        float* dng = denom + (size_t)g * N * HEADS;

        gemm_xp_kernel<<<mtiles, 128, 0, stream>>>(x, Wh + (size_t)g * 64 * 64, xpg, N);
        logits_kernel<<<nblkN, 256, 0, stream>>>(xpg, asg[g], adg[g], asg_n, adg_n, N);
        init_max_kernel<<<nblkN, 256, 0, stream>>>(asg_n, adg_n, mxg, N);
        edge_max_kernel<<<nblkE, 256, 0, stream>>>(ei[g], asg_n, adg_n, mxg, E);
        node_denom_kernel<<<nblkN, 256, 0, stream>>>(asg_n, adg_n, mxg, dng, N);
        edge_denom_kernel<<<nblkE, 256, 0, stream>>>(ei[g], asg_n, adg_n, mxg, dng, E);
        node_selfmsg_kernel<<<(N * 64 + 255) / 256, 256, 0, stream>>>(xpg, asg_n, adg_n,
                                                                      mxg, dng, agr, N);
        {   // 32 lanes per edge
            long long tot = (long long)E * 32;
            int blocks = (int)((tot + 255) / 256);
            edge_msg_kernel<<<blocks, 256, 0, stream>>>(ei[g], xpg, asg_n, adg_n, mxg,
                                                        dng, agr, E);
        }
    }

    combine_kernel<<<(N * HID + 255) / 256, 256, 0, stream>>>(
        agg, agg + (size_t)N * 64, agg + (size_t)2 * N * 64,
        bg[0], bg[1], bg[2], alph[0], alph[1], alph[2], xc, N);

    kan_kernel<<<mtiles, 64, 0, stream>>>(xc, XC_STRIDE, FUSE_IN, grid1, We1, KP1, h1, 0, N);
    kan_kernel<<<mtiles, 64, 0, stream>>>(h1, HID, HID, grid2, We2, KP2, (float*)d_out, 1, N);
}
